// CrissCrossAttention_78958678769766
// MI455X (gfx1250) — matrix operand order repaired and checked by reading, not compiled, not run on hardware
//
#include <hip/hip_runtime.h>
#include <hip/hip_bf16.h>
#include <cmath>
#include <cstdint>

// ---------------------------------------------------------------------------
// EEG transformer forward on gfx1250 (wave32, WMMA, async LDS staging).
// All GEMMs use v_wmma_f32_16x16x32_f16 (f16 in, f32 accumulate).
// Tiles are staged to LDS as raw f32 via global_load_async_to_lds_b128 when
// available (ASYNCcnt path); conversion to f16 happens at fragment build.
// ---------------------------------------------------------------------------

typedef __attribute__((ext_vector_type(16))) _Float16 v16h;
typedef __attribute__((ext_vector_type(8)))  float    v8f;
typedef int i128 __attribute__((vector_size(16)));   // builtin's V4i type

#define DB 2
#define DC 64
#define DT 512
#define DH 128
#define NHEADS 4
#define HDIM 32
#define NTOK (DB * DT * DC)             // 65536 tokens
#define ATT_SCALE 0.17677669529663689f  // 1/sqrt(32)

#if defined(__HIP_DEVICE_COMPILE__) && \
    __has_builtin(__builtin_amdgcn_global_load_async_to_lds_b128)
#define USE_ASYNC_LDS 1
#else
#define USE_ASYNC_LDS 0
#endif

#if USE_ASYNC_LDS
__device__ __forceinline__ void async_cp16(const float* g, float* l) {
  // 16-byte global -> LDS async copy (no VGPR round trip).
  __builtin_amdgcn_global_load_async_to_lds_b128(
      (__attribute__((address_space(1))) i128*)(uintptr_t)(const void*)g,
      (__attribute__((address_space(3))) i128*)(uint32_t)(uintptr_t)(void*)l,
      0, 0);
}
__device__ __forceinline__ void async_wait0() {
#if __has_builtin(__builtin_amdgcn_s_wait_asynccnt)
  __builtin_amdgcn_s_wait_asynccnt(0);
#else
  asm volatile("s_wait_asynccnt 0" ::: "memory");
#endif
}
#endif

// 16B tile-chunk copy: async DMA when available, else b128 load/store pair.
__device__ __forceinline__ void stage16(const float* g, float* l) {
#if USE_ASYNC_LDS
  async_cp16(g, l);
#else
  *(float4*)l = *(const float4*)g;
#endif
}
__device__ __forceinline__ void stage_fence() {
#if USE_ASYNC_LDS
  async_wait0();
#endif
  __syncthreads();
}

// ---------------- WMMA helpers -------------------------------------------

__device__ __forceinline__ v8f wmma16(v16h a, v16h b, v8f c) {
  return __builtin_amdgcn_wmma_f32_16x16x32_f16(
      /*neg_a=*/false, a, /*neg_b=*/false, b,
      /*c_mod=*/(short)0, c, /*reuse_a=*/false, /*reuse_b=*/false);
}

// A fragment (16x32) from row-major *f16* LDS tile.
// Lanes 0-15: rows M=lane, K in {0..7,16..23}; lanes 16-31: K {8..15,24..31}.
__device__ __forceinline__ v16h frag_a_h(const _Float16* A, int stride, int row,
                                         int kOff, int lane) {
  const _Float16* p = A + row * stride + kOff + ((lane >> 4) ? 8 : 0);
  v16h a;
#pragma unroll
  for (int q = 0; q < 8; ++q) { a[q] = p[q]; a[8 + q] = p[16 + q]; }
  return a;
}

// A fragment (16x32) from row-major *f32* LDS tile (convert at build).
__device__ __forceinline__ v16h frag_a_f32(const float* A, int stride, int row,
                                           int kOff, int lane) {
  const float* p = A + row * stride + kOff + ((lane >> 4) ? 8 : 0);
  v16h a;
#pragma unroll
  for (int q = 0; q < 8; ++q) {
    a[q] = (_Float16)p[q];
    a[8 + q] = (_Float16)p[16 + q];
  }
  return a;
}

// B fragment (32x16) from *transposed* f32 tile Bt[n][k] (contiguous in k).
__device__ __forceinline__ v16h frag_bt_f32(const float* Bt, int stride,
                                            int ncol0, int kOff, int lane) {
  const float* p = Bt + (ncol0 + (lane & 15)) * stride + kOff +
                   ((lane >> 4) ? 8 : 0);
  v16h b;
#pragma unroll
  for (int q = 0; q < 16; ++q) b[q] = (_Float16)p[q < 8 ? q : q + 8];
  return b;
}

// B fragment (32x16) from row-major (k-major) f32 tile Wk[k][n].
// Lanes 0-15: K={0..7,16..23}+kOff of column n; lanes 16-31: K={8..15,24..31}+kOff.
__device__ __forceinline__ v16h frag_b_km(const float* Wk, int stride,
                                          int ncol0, int kOff, int lane) {
  const float* p = Wk + (kOff + ((lane >> 4) ? 8 : 0)) * stride + ncol0 +
                   (lane & 15);
  v16h b;
#pragma unroll
  for (int q = 0; q < 16; ++q) b[q] = (_Float16)p[(q < 8 ? q : q + 8) * stride];
  return b;
}

__device__ __forceinline__ float gelu_f(float x) {
  return 0.5f * x * (1.0f + erff(x * 0.70710678118654752f));
}

// ---------------- embed: input proj + positional encodings ----------------
__global__ void embed_kernel(const float* __restrict__ eeg,
                             const float* __restrict__ wb,
                             const float* __restrict__ ww,
                             const float* __restrict__ spos,
                             const float* __restrict__ tpos,
                             float* __restrict__ h) {
  int token = blockIdx.x;
  int c = token % DC;
  int t = (token / DC) % DT;
  int b = token / (DC * DT);
  int j = threadIdx.x;
  float x = eeg[((size_t)b * DC + c) * DT + t];
  h[(size_t)token * DH + j] =
      x * ww[j] + wb[j] + spos[c * DH + j] + tpos[t * DH + j];
}

// ---------------- generic linear: Out[M,N] = act(A[M,K] @ W[K,N] + bias) ---
// block = 256 threads = 8 waves (4 along M x 2 along N); block tile 64x64.
__global__ void linear_kernel(const float* __restrict__ A,
                              const float* __restrict__ W,
                              const float* __restrict__ bias,
                              float* __restrict__ Out,
                              int M, int N, int K, int act) {
  __shared__ float Asf[64 * 32];  // A tile, row-major          (8 KB)
  __shared__ float Wsf[32 * 64];  // W tile, k-major [k][n]     (8 KB)

  int tid = threadIdx.x;
  int lane = tid & 31, wid = tid >> 5;
  int wm = wid >> 1, wn = wid & 1;
  int m0 = blockIdx.x * 64, n0 = blockIdx.y * 64;

  v8f acc0 = {}; v8f acc1 = {};

  for (int k0 = 0; k0 < K; k0 += 32) {
    // A tile: 64x32 f32 = 512 x 16B chunks
#pragma unroll
    for (int cc = 0; cc < 2; ++cc) {
      int c = tid + cc * 256;
      int r = c >> 3, kq = (c & 7) * 4;
      const float* gp = A + (size_t)(m0 + r) * K + k0 + kq;
      if (k0 + 32 < K) __builtin_prefetch(gp + 32, 0, 1);
      stage16(gp, &Asf[r * 32 + kq]);
    }
    // W tile: 32x64 f32 (rows contiguous in n)
#pragma unroll
    for (int cc = 0; cc < 2; ++cc) {
      int c = tid + cc * 256;
      int k = c >> 4, nq = (c & 15) * 4;
      stage16(W + (size_t)(k0 + k) * N + n0 + nq, &Wsf[k * 64 + nq]);
    }
    stage_fence();

    v16h a  = frag_a_f32(Asf, 32, wm * 16 + (lane & 15), 0, lane);
    v16h b0 = frag_b_km(Wsf, 64, wn * 32 + 0, 0, lane);
    v16h b1 = frag_b_km(Wsf, 64, wn * 32 + 16, 0, lane);
    acc0 = wmma16(a, b0, acc0);
    acc1 = wmma16(a, b1, acc1);
    __syncthreads();
  }

  int hi = lane >> 4;
  int rowb = m0 + wm * 16 + 8 * hi;
  int col0 = n0 + wn * 32 + (lane & 15);
#pragma unroll
  for (int rr = 0; rr < 8; ++rr) {
    float v0 = acc0[rr] + bias[col0];
    float v1 = acc1[rr] + bias[col0 + 16];
    if (act == 1) { v0 = gelu_f(v0); v1 = gelu_f(v1); }
    Out[(size_t)(rowb + rr) * N + col0] = v0;
    Out[(size_t)(rowb + rr) * N + col0 + 16] = v1;
  }
}

// ---------------- attention (flash-style, online softmax) -----------------
// Q,K,V,O: [Bn, L, DH]; head slice = columns hh*32..+31.
// grid.x = Bn*NHEADS, grid.y = L/64. block = 128 (4 waves), 16 q-rows/wave.
__global__ void attn_kernel(const float* __restrict__ Q,
                            const float* __restrict__ K,
                            const float* __restrict__ V,
                            float* __restrict__ O, int L) {
  __shared__ float Ksf[64 * 32];       // K rows [key][dim]  (Bt for Q@K^T)
  __shared__ float Vsf[64 * 32];       // V rows [key][dim]  (k-major for P@V)
  __shared__ _Float16 Ps[4][16 * 64];  // per-wave P tile (f16)

  int tid = threadIdx.x;
  int lane = tid & 31, wid = tid >> 5;
  int hi = lane >> 4, lo = lane & 15;
  int b  = blockIdx.x / NHEADS;
  int hh = blockIdx.x % NHEADS;
  int q0 = blockIdx.y * 64;

  // Load Q fragment once (softmax scale folded in).
  int rowg = q0 + wid * 16 + lo;
  const float* qp = Q + ((size_t)b * L + rowg) * DH + hh * HDIM;
  int kb = hi ? 8 : 0;
  v16h qf;
#pragma unroll
  for (int q = 0; q < 8; ++q) {
    qf[q]     = (_Float16)(qp[kb + q] * ATT_SCALE);
    qf[8 + q] = (_Float16)(qp[kb + 16 + q] * ATT_SCALE);
  }

  float mrow[8], lrow[8];
#pragma unroll
  for (int r = 0; r < 8; ++r) { mrow[r] = -3.0e38f; lrow[r] = 0.0f; }
  v8f acc0 = {}; v8f acc1 = {};

  for (int j0 = 0; j0 < L; j0 += 64) {
    // stage K and V blocks (each 64 keys x 32 dims f32, 512 x 16B chunks)
#pragma unroll
    for (int cc = 0; cc < 4; ++cc) {
      int c = tid + cc * 128;
      int r = c >> 3, kq = (c & 7) * 4;
      size_t rowbase = ((size_t)b * L + j0 + r) * DH + hh * HDIM + kq;
      stage16(K + rowbase, &Ksf[r * 32 + kq]);
      stage16(V + rowbase, &Vsf[r * 32 + kq]);
    }
    stage_fence();

    // scores S = (Q*scale) @ K^T : four 16x16 tiles along the 64 keys
    v8f s[4];
#pragma unroll
    for (int t = 0; t < 4; ++t) {
      v8f z = {};
      s[t] = wmma16(qf, frag_bt_f32(Ksf, 32, t * 16, 0, lane), z);
    }

    // online softmax update (stats replicated across each 16-lane half)
#pragma unroll
    for (int r = 0; r < 8; ++r) {
      float mx = fmaxf(fmaxf(s[0][r], s[1][r]), fmaxf(s[2][r], s[3][r]));
#pragma unroll
      for (int o = 1; o < 16; o <<= 1) mx = fmaxf(mx, __shfl_xor(mx, o, 32));
      float mnew = fmaxf(mrow[r], mx);
      float corr = __expf(mrow[r] - mnew);
      mrow[r] = mnew;
      float p0 = __expf(s[0][r] - mnew);
      float p1 = __expf(s[1][r] - mnew);
      float p2 = __expf(s[2][r] - mnew);
      float p3 = __expf(s[3][r] - mnew);
      float rs = p0 + p1 + p2 + p3;
#pragma unroll
      for (int o = 1; o < 16; o <<= 1) rs += __shfl_xor(rs, o, 32);
      lrow[r] = lrow[r] * corr + rs;
      acc0[r] *= corr;
      acc1[r] *= corr;
      _Float16* pp = &Ps[wid][(r + 8 * hi) * 64 + lo];
      pp[0]  = (_Float16)p0;
      pp[16] = (_Float16)p1;
      pp[32] = (_Float16)p2;
      pp[48] = (_Float16)p3;
    }
    __syncthreads();

    // acc += P @ V  (K-dim = 64 keys, two chunks of 32)
#pragma unroll
    for (int kc = 0; kc < 2; ++kc) {
      v16h pa = frag_a_h(Ps[wid], 64, lo, kc * 32, lane);
      acc0 = wmma16(pa, frag_b_km(Vsf, 32, 0,  kc * 32, lane), acc0);
      acc1 = wmma16(pa, frag_b_km(Vsf, 32, 16, kc * 32, lane), acc1);
    }
    __syncthreads();
  }

  // normalize + store
#pragma unroll
  for (int r = 0; r < 8; ++r) {
    float inv = 1.0f / lrow[r];
    int rg = q0 + wid * 16 + r + 8 * hi;
    float* op = O + ((size_t)b * L + rg) * DH + hh * HDIM + lo;
    op[0]  = acc0[r] * inv;
    op[16] = acc1[r] * inv;
  }
}

// ---------------- fused FFN: Out = gelu(X@W1+b1)@W2+b2 --------------------
// 32 tokens per block; 512-wide intermediate stays in LDS (f16).
__global__ void ffn_kernel(const float* __restrict__ X,
                           const float* __restrict__ W1,
                           const float* __restrict__ B1,
                           const float* __restrict__ W2,
                           const float* __restrict__ B2,
                           float* __restrict__ Out) {
  __shared__ float Xsf[32 * 128];     // 16 KB (f32, async-staged)
  __shared__ _Float16 Fs[32 * 512];   // 32 KB (intermediate, f16)
  __shared__ float Wsf[32 * 64];      //  8 KB (weight tile, k-major f32)

  int tid = threadIdx.x;
  int lane = tid & 31, wid = tid >> 5;
  int hi = lane >> 4, lo = lane & 15;
  int wm = wid >> 2, wn = wid & 3;    // 2 x 4 wave grid
  int m0 = blockIdx.x * 32;

  // stage X tile (32 x 128 f32 = 1024 x 16B chunks)
#pragma unroll
  for (int cc = 0; cc < 4; ++cc) {
    int c = tid + cc * 256;
    int r = c >> 5, kq = (c & 31) * 4;
    stage16(X + (size_t)(m0 + r) * DH + kq, &Xsf[r * 128 + kq]);
  }
  stage_fence();

  // stage 1: F = gelu(X @ W1 + b1), 64 output cols per pass
  for (int n0 = 0; n0 < 512; n0 += 64) {
    v8f acc = {};
    for (int k0 = 0; k0 < 128; k0 += 32) {
#pragma unroll
      for (int cc = 0; cc < 2; ++cc) {
        int c = tid + cc * 256;
        int k = c >> 4, nq = (c & 15) * 4;
        stage16(W1 + (size_t)(k0 + k) * 512 + n0 + nq, &Wsf[k * 64 + nq]);
      }
      stage_fence();
      v16h a = frag_a_f32(Xsf, 128, wm * 16 + lo, k0, lane);
      acc = wmma16(a, frag_b_km(Wsf, 64, wn * 16, 0, lane), acc);
      __syncthreads();
    }
    int col = n0 + wn * 16 + lo;
#pragma unroll
    for (int rr = 0; rr < 8; ++rr) {
      float v = gelu_f(acc[rr] + B1[col]);
      Fs[(wm * 16 + rr + 8 * hi) * 512 + col] = (_Float16)v;
    }
  }
  __syncthreads();

  // stage 2: Out = F @ W2 + b2
  for (int n0 = 0; n0 < 128; n0 += 64) {
    v8f acc = {};
    for (int k0 = 0; k0 < 512; k0 += 32) {
#pragma unroll
      for (int cc = 0; cc < 2; ++cc) {
        int c = tid + cc * 256;
        int k = c >> 4, nq = (c & 15) * 4;
        stage16(W2 + (size_t)(k0 + k) * 128 + n0 + nq, &Wsf[k * 64 + nq]);
      }
      stage_fence();
      v16h a = frag_a_h(Fs, 512, wm * 16 + lo, k0, lane);
      acc = wmma16(a, frag_b_km(Wsf, 64, wn * 16, 0, lane), acc);
      __syncthreads();
    }
    int col = n0 + wn * 16 + lo;
#pragma unroll
    for (int rr = 0; rr < 8; ++rr)
      Out[(size_t)(m0 + wm * 16 + rr + 8 * hi) * DH + col] = acc[rr] + B2[col];
  }
}

// ---------------- transpose: in[b][d1][d2][H] -> out[b][d2][d1][H] --------
__global__ void transpose_kernel(const float* __restrict__ in,
                                 float* __restrict__ out, int D1, int D2) {
  int idx = blockIdx.x;
  int d2 = idx % D2;
  int d1 = (idx / D2) % D1;
  int b = idx / (D1 * D2);
  int j = threadIdx.x;
  out[(((size_t)b * D2 + d2) * D1 + d1) * DH + j] =
      in[(((size_t)b * D1 + d1) * D2 + d2) * DH + j];
}

// ---------------- residual add + LayerNorm (one wave per token) -----------
__global__ void add_ln_kernel(const float* __restrict__ X,
                              const float* __restrict__ Y,
                              float* __restrict__ O,
                              const float* __restrict__ g,
                              const float* __restrict__ beta, int transposed) {
  int token = blockIdx.x;
  int lane = threadIdx.x;
  int c = token % DC;
  int t = (token / DC) % DT;
  int b = token / (DC * DT);
  size_t xbase = (size_t)token * DH;
  size_t ybase = transposed ? (((size_t)b * DC + c) * DT + t) * DH : xbase;

  float v[4];
  float s = 0.0f;
#pragma unroll
  for (int e = 0; e < 4; ++e) {
    int j = lane * 4 + e;
    v[e] = X[xbase + j] + Y[ybase + j];
    s += v[e];
  }
#pragma unroll
  for (int o = 1; o < 32; o <<= 1) s += __shfl_xor(s, o, 32);
  float mu = s * (1.0f / DH);
  float sq = 0.0f;
#pragma unroll
  for (int e = 0; e < 4; ++e) {
    float d = v[e] - mu;
    sq += d * d;
  }
#pragma unroll
  for (int o = 1; o < 32; o <<= 1) sq += __shfl_xor(sq, o, 32);
  float rs = rsqrtf(sq * (1.0f / DH) + 1e-5f);
#pragma unroll
  for (int e = 0; e < 4; ++e) {
    int j = lane * 4 + e;
    O[xbase + j] = (v[e] - mu) * rs * g[j] + beta[j];
  }
}

// ---------------- output projection: out[b][c][t] = h . w + b -------------
__global__ void outproj_kernel(const float* __restrict__ h,
                               const float* __restrict__ w,
                               const float* __restrict__ bias,
                               float* __restrict__ out) {
  int token = blockIdx.x;
  int lane = threadIdx.x;
  int c = token % DC;
  int t = (token / DC) % DT;
  int b = token / (DC * DT);
  float s = 0.0f;
#pragma unroll
  for (int e = 0; e < 4; ++e) {
    int j = lane * 4 + e;
    s += h[(size_t)token * DH + j] * w[j];
  }
#pragma unroll
  for (int o = 1; o < 32; o <<= 1) s += __shfl_xor(s, o, 32);
  if (lane == 0) out[((size_t)b * DC + c) * DT + t] = s + bias[0];
}

// ---------------------------------------------------------------------------
// Orchestration. Input leaf order = JAX sorted-dict pytree flatten:
// 0 eeg; 1 input_proj.b; 2 input_proj.w;
// per layer (base=3+26*l): 0 ffn.w1.b 1 ffn.w1.w 2 ffn.w2.b 3 ffn.w2.w
//   4 ffn_norm.b 5 ffn_norm.g
//   6 sp.k.b 7 sp.k.w 8 sp.norm.b 9 sp.norm.g 10 sp.o.b 11 sp.o.w
//   12 sp.q.b 13 sp.q.w 14 sp.v.b 15 sp.v.w ; 16..25 temporal same pattern;
// 55 output_proj.b; 56 output_proj.w; 57 spatial_pos; 58 temporal_pos.
// ---------------------------------------------------------------------------
extern "C" void kernel_launch(void* const* d_in, const int* in_sizes, int n_in,
                              void* d_out, int out_size, void* d_ws,
                              size_t ws_size, hipStream_t stream) {
  if (n_in < 59) return;
  auto in = [&](int i) { return (const float*)d_in[i]; };

  const float* eeg  = in(0);
  const float* ipb  = in(1);
  const float* ipw  = in(2);
  const float* opb  = in(55);
  const float* opw  = in(56);
  const float* spos = in(57);
  const float* tpos = in(58);

  const size_t SZ = (size_t)NTOK * DH * sizeof(float);
  char* w8 = (char*)d_ws;
  float* h  = (float*)(w8 + 0 * SZ);
  float* bA = (float*)(w8 + 1 * SZ);
  float* bB = (float*)(w8 + 2 * SZ);
  float* bC = (float*)(w8 + 3 * SZ);
  float* bD = (float*)(w8 + 4 * SZ);

  auto linear = [&](const float* A, const float* W, const float* b, float* O,
                    int M, int N, int K, int act) {
    linear_kernel<<<dim3(M / 64, N / 64), 256, 0, stream>>>(A, W, b, O, M, N,
                                                            K, act);
  };

  embed_kernel<<<NTOK, DH, 0, stream>>>(eeg, ipb, ipw, spos, tpos, h);

  for (int l = 0; l < 2; ++l) {
    int base = 3 + 26 * l;
    // ---- spatial attention (x = h viewed [B*T, C, H], contiguous) ----
    linear(h, in(base + 13), in(base + 12), bA, NTOK, DH, DH, 0);  // Q
    linear(h, in(base + 7),  in(base + 6),  bB, NTOK, DH, DH, 0);  // K
    linear(h, in(base + 15), in(base + 14), bC, NTOK, DH, DH, 0);  // V
    attn_kernel<<<dim3(DB * DT * NHEADS, DC / 64), 128, 0, stream>>>(
        bA, bB, bC, bD, DC);
    linear(bD, in(base + 11), in(base + 10), bA, NTOK, DH, DH, 0); // O-proj
    add_ln_kernel<<<NTOK, 32, 0, stream>>>(h, bA, h, in(base + 9),
                                           in(base + 8), 0);

    // ---- temporal attention (x = h transposed to [B*C, T, H]) ----
    transpose_kernel<<<NTOK, DH, 0, stream>>>(h, bD, DT, DC);
    linear(bD, in(base + 23), in(base + 22), bA, NTOK, DH, DH, 0); // Q
    linear(bD, in(base + 17), in(base + 16), bB, NTOK, DH, DH, 0); // K
    linear(bD, in(base + 25), in(base + 24), bC, NTOK, DH, DH, 0); // V
    attn_kernel<<<dim3(DB * DC * NHEADS, DT / 64), 128, 0, stream>>>(
        bA, bB, bC, bD, DT);
    linear(bD, in(base + 21), in(base + 20), bA, NTOK, DH, DH, 0); // O-proj
    add_ln_kernel<<<NTOK, 32, 0, stream>>>(h, bA, h, in(base + 19),
                                           in(base + 18), 1);

    // ---- fused FFN ----
    ffn_kernel<<<NTOK / 32, 256, 0, stream>>>(h, in(base + 1), in(base + 0),
                                              in(base + 3), in(base + 2), bA);
    add_ln_kernel<<<NTOK, 32, 0, stream>>>(h, bA, h, in(base + 5),
                                           in(base + 4), 0);
  }

  outproj_kernel<<<NTOK, 32, 0, stream>>>(h, opw, opb, (float*)d_out);
}
